// Decoder_22720376996057
// MI455X (gfx1250) — matrix operand order repaired and checked by reading, not compiled, not run on hardware
//
#include <hip/hip_runtime.h>

#define NF   64
#define FIN  192
#define IMG  64
#define HW   4096
#define BSZ  16
#define LDIM 512
#define NCALLS 16
#define EPSV 1e-5f
#define CSTR 200   // padded LDS stride (halves) per pixel row: 400B -> conflict-free b128 reads

typedef __attribute__((ext_vector_type(16))) _Float16 v16h;
typedef __attribute__((ext_vector_type(8)))  _Float16 v8h;
typedef __attribute__((ext_vector_type(8)))  float    v8f;

// ---------------- hypernetwork weights: w = lat@Wk + bk ----------------
__global__ void k_weights(const float* __restrict__ lat, const float* __restrict__ Wk,
                          const float* __restrict__ bk, float* __restrict__ w) {
  int idx = blockIdx.x * blockDim.x + threadIdx.x;      // 0 .. 16*12288-1
  int b = idx / (NF * FIN);
  int j = idx - b * (NF * FIN);
  const float* latb = lat + b * LDIM;
  float acc = bk[j];
  for (int l = 0; l < LDIM; ++l) acc += latb[l] * Wk[l * (NF * FIN) + j];
  w[idx] = acc;
}

__global__ void k_bvec(const float* __restrict__ lat, const float* __restrict__ Wb,
                       const float* __restrict__ bb, float* __restrict__ bvec) {
  int idx = blockIdx.x * blockDim.x + threadIdx.x;      // 0 .. 1023
  int b = idx / NF, o = idx - b * NF;
  float acc = bb[o];
  for (int l = 0; l < LDIM; ++l) acc += lat[b * LDIM + l] * Wb[l * NF + o];
  bvec[idx] = acc;
}

// ---------------- init: embs[0] = out0 ----------------
__global__ void k_init(float* __restrict__ e0) {
  int idx = blockIdx.x * blockDim.x + threadIdx.x;      // 0 .. B*NF*HW/4-1 (4096 blocks x 256)
  int r = idx % (NF * HW);                              // c*HW + px
  e0[idx] = (r == (32 * IMG + 32)) ? 1.0f : 0.0f;       // c==0, center pixel
}

// ---------------- sobel helpers (lax conv = cross-correlation, SAME/zero pad) ----
__device__ __forceinline__ float ldz(const float* ch, int y, int x) {
  if ((unsigned)y >= IMG || (unsigned)x >= IMG) return 0.f;
  return ch[y * IMG + x];
}
__device__ __forceinline__ float pval(const float* ch, int f, int y, int x) {
  if (f == 0) return ch[y * IMG + x];
  if (f == 1)  // kx = [[-1,0,1],[-2,0,2],[-1,0,1]]/8
    return (-ldz(ch,y-1,x-1) + ldz(ch,y-1,x+1)
            -2.f*ldz(ch,y,x-1) + 2.f*ldz(ch,y,x+1)
            -ldz(ch,y+1,x-1) + ldz(ch,y+1,x+1)) * 0.125f;
  // ky = kx.T
  return (-ldz(ch,y-1,x-1) - 2.f*ldz(ch,y-1,x) - ldz(ch,y-1,x+1)
          +ldz(ch,y+1,x-1) + 2.f*ldz(ch,y+1,x) + ldz(ch,y+1,x+1)) * 0.125f;
}

// ---------------- per-(b, p-channel) mean / rstd ----------------
__global__ void k_stats(const float* __restrict__ state,
                        float* __restrict__ mu, float* __restrict__ rs) {
  int pc = blockIdx.x;   // 0..191
  int b  = blockIdx.y;   // 0..15
  int f = pc / NF, ch = pc - f * NF;
  const float* chp = state + (size_t)(b * NF + ch) * HW;
  float s1 = 0.f, s2 = 0.f;
  for (int px = threadIdx.x; px < HW; px += 256) {
    float v = pval(chp, f, px >> 6, px & 63);
    s1 += v; s2 += v * v;
  }
  __shared__ float r1[256], r2[256];
  r1[threadIdx.x] = s1; r2[threadIdx.x] = s2;
  __syncthreads();
  for (int s = 128; s > 0; s >>= 1) {
    if (threadIdx.x < s) { r1[threadIdx.x] += r1[threadIdx.x + s]; r2[threadIdx.x] += r2[threadIdx.x + s]; }
    __syncthreads();
  }
  if (threadIdx.x == 0) {
    float m   = r1[0] * (1.f / HW);
    float var = r2[0] * (1.f / HW) - m * m;
    mu[b * FIN + pc] = m;
    rs[b * FIN + pc] = rsqrtf(var + EPSV);
  }
}

// ---------------- fold norm into weights: w' = w*rs (f16), bias' = b - sum(w*rs*mu) ----
__global__ void k_prep(const float* __restrict__ w, const float* __restrict__ bvec,
                       const float* __restrict__ mu, const float* __restrict__ rs,
                       _Float16* __restrict__ wp, float* __restrict__ biasp) {
  int idx = blockIdx.x * blockDim.x + threadIdx.x;  // 0..1023  (b,o)
  int b = idx / NF;
  const float* wrow = w  + (size_t)idx * FIN;
  const float* mub  = mu + b * FIN;
  const float* rsb  = rs + b * FIN;
  _Float16* wpo = wp + (size_t)idx * FIN;
  float acc = 0.f;
  for (int c = 0; c < FIN; ++c) {
    float wsv = wrow[c] * rsb[c];
    wpo[c] = (_Float16)wsv;
    acc += wsv * mub[c];
  }
  biasp[idx] = bvec[idx] - acc;
}

// ---------------- fused step GEMM: stage norm-f16 p tile in LDS, WMMA, leak-update ----
__global__ __launch_bounds__(256) void k_gemm(
    const float* __restrict__ state,        // embs[t]   (B,NF,HW)
    const _Float16* __restrict__ wp,        // (B,NF,FIN) f16
    const float* __restrict__ biasp,        // (B,NF)
    const float* __restrict__ mu, const float* __restrict__ rs,
    const float* __restrict__ leakp,
    float* __restrict__ next)               // embs[t+1]
{
  // pixel-major p tile: plds[px][c] so each lane's 16-K B fragment is contiguous
  __shared__ _Float16 plds[128 * CSTR];     // 128 px x 192 ch (padded) = 51.2 KB
  int pb = blockIdx.x;                      // 32 pixel blocks (2 rows each)
  int b  = blockIdx.y;
  int tid = threadIdx.x;
  int y0 = pb * 2;
  const float* sb  = state + (size_t)b * NF * HW;
  const float* mub = mu + b * FIN;
  const float* rsb = rs + b * FIN;

  // stage normalized p (f16): 192 channels x 128 pixels (transposed store)
  for (int idx = tid; idx < FIN * 128; idx += 256) {
    int c  = idx >> 7;
    int px = idx & 127;
    int f = c / NF, ch = c - f * NF;
    float v = pval(sb + (size_t)ch * HW, f, y0 + (px >> 6), px & 63);
    plds[px * CSTR + c] = (_Float16)((v - mub[c]) * rsb[c]);
  }
  __syncthreads();

  int wave = tid >> 5, lane = tid & 31;
  int half = lane >> 4;          // 0: lanes 0-15, 1: lanes 16-31
  int lidx = lane & 15;
  int koffA = half * 16;         // A f16 16x32 per-lane K split {0-15}/{16-31}
  int koffB = half * 16;         // B f16 32x16 per-lane K split {0-15}/{16-31}
  int pxl = wave * 16 + lidx;    // this lane's pixel column (N index)
  const _Float16* wpb = wp + (size_t)b * NF * FIN;
  const _Float16* brow = plds + pxl * CSTR;

  v8f acc[4] = {v8f{}, v8f{}, v8f{}, v8f{}};

  for (int kc = 0; kc < FIN; kc += 32) {
    // B fragment: 16 consecutive K halves -> two ds_load_b128
    const _Float16* bp = brow + kc + koffB;
    v8h b0 = *(const v8h*)(bp);
    v8h b1 = *(const v8h*)(bp + 8);
    v16h bf;
    #pragma unroll
    for (int e = 0; e < 8; ++e) { bf[e] = b0[e]; bf[8 + e] = b1[e]; }

    #pragma unroll
    for (int mt = 0; mt < 4; ++mt) {
      int o = mt * 16 + lidx;   // A row = output channel
      const _Float16* row = wpb + (size_t)o * FIN + kc + koffA;
      v8h lo = *(const v8h*)(row);
      v8h hi = *(const v8h*)(row + 8);
      v16h af;
      #pragma unroll
      for (int e = 0; e < 8; ++e) { af[e] = lo[e]; af[8 + e] = hi[e]; }
      acc[mt] = __builtin_amdgcn_wmma_f32_16x16x32_f16(
          false, af, false, bf, (short)0, acc[mt], false, false);
    }
  }

  float lk = fminf(fmaxf(leakp[0], 0.001f), 1000.f);
  int pxg = pb * 128 + pxl;                 // D: N = lane%16
  float* nb = next + (size_t)b * NF * HW;
  #pragma unroll
  for (int mt = 0; mt < 4; ++mt) {
    #pragma unroll
    for (int i = 0; i < 8; ++i) {
      int o = mt * 16 + i + half * 8;       // D: M = vgpr + 8*(lane/16)
      float nv = acc[mt][i] + biasp[b * NF + o];
      nb[(size_t)o * HW + pxg] = sb[(size_t)o * HW + pxg] + lk * nv;
    }
  }
}

// ---------------- finalize: clip + raw views of embs[16][:, :3] ----------------
__global__ void k_final(const float* __restrict__ e16,
                        float* __restrict__ outc, float* __restrict__ outr) {
  int idx = blockIdx.x * blockDim.x + threadIdx.x;   // 0 .. 16*3*4096-1
  int b = idx / (3 * HW);
  int r = idx - b * (3 * HW);
  float v = e16[(size_t)b * NF * HW + r];
  outr[idx] = v;
  outc[idx] = fminf(fmaxf(v, -1.f), 1.f);
}

extern "C" void kernel_launch(void* const* d_in, const int* in_sizes, int n_in,
                              void* d_out, int out_size, void* d_ws, size_t ws_size,
                              hipStream_t stream) {
  const float* lat  = (const float*)d_in[0];
  const float* Wk   = (const float*)d_in[1];
  const float* bk   = (const float*)d_in[2];
  const float* Wb   = (const float*)d_in[3];
  const float* bb   = (const float*)d_in[4];
  const float* leak = (const float*)d_in[5];

  float* out   = (float*)d_out;
  float* clipd = out;                                  // (16,3,64,64)
  float* embs  = out + BSZ * 3 * HW;                   // (17,16,64,64,64) = state history
  float* raw   = embs + (size_t)(NCALLS + 1) * BSZ * NF * HW;  // (16,3,64,64)

  char* wsb = (char*)d_ws;
  float*    w     = (float*)wsb;     wsb += (size_t)BSZ * NF * FIN * 4;  // 786432 B
  float*    bvec  = (float*)wsb;     wsb += BSZ * NF * 4;
  float*    mu    = (float*)wsb;     wsb += BSZ * FIN * 4;
  float*    rs    = (float*)wsb;     wsb += BSZ * FIN * 4;
  float*    biasp = (float*)wsb;     wsb += BSZ * NF * 4;
  _Float16* wp    = (_Float16*)wsb;  wsb += (size_t)BSZ * NF * FIN * 2;

  k_weights<<<768, 256, 0, stream>>>(lat, Wk, bk, w);
  k_bvec<<<4, 256, 0, stream>>>(lat, Wb, bb, bvec);
  k_init<<<4096, 256, 0, stream>>>(embs);

  const size_t stateN = (size_t)BSZ * NF * HW;
  for (int t = 0; t < NCALLS; ++t) {
    float* cur = embs + (size_t)t * stateN;
    float* nxt = cur + stateN;
    k_stats<<<dim3(FIN, BSZ), 256, 0, stream>>>(cur, mu, rs);
    k_prep<<<4, 256, 0, stream>>>(w, bvec, mu, rs, wp, biasp);
    k_gemm<<<dim3(32, BSZ), 256, 0, stream>>>(cur, wp, biasp, mu, rs, leak, nxt);
  }
  k_final<<<768, 256, 0, stream>>>(embs + (size_t)NCALLS * stateN, clipd, raw);
}
